// FUB_2757369004556
// MI455X (gfx1250) — compile-verified, matrix operand order checked by reading, not run on hardware
//
#include <hip/hip_runtime.h>
#include <hip/hip_bf16.h>
#include <math.h>

#define EPS 1e-5f

typedef __attribute__((ext_vector_type(8)))  __bf16 v8bf;
typedef __attribute__((ext_vector_type(16))) __bf16 v16bf;
typedef __attribute__((ext_vector_type(8)))  float  v8f;

#define XSTRIDE 200   // bf16 units; 400 B rows, 16B aligned, dword stride 100
#define HSTRIDE 104   // bf16 units; 208 B rows, 16B aligned, dword stride 52

// Build a 16x32 bf16 A/B fragment half for this lane: two 16-byte loads.
// off = k0 + (lane>>4)*8 ; second half of the lane's K values live +16 further.
static __device__ inline v16bf load_frag(const __bf16* base, int off) {
  v8bf lo = *reinterpret_cast<const v8bf*>(base + off);
  v8bf hi = *reinterpret_cast<const v8bf*>(base + off + 16);
  return __builtin_shufflevector(lo, hi, 0,1,2,3,4,5,6,7,8,9,10,11,12,13,14,15);
}

// ---------------- Kernel 1: global average pool  gap[b][ch] ----------------
__global__ __launch_bounds__(256)
void gap_kernel(const float* __restrict__ x1, const float* __restrict__ x2,
                const float* __restrict__ x3, float* __restrict__ gap) {
  int blk = blockIdx.x;              // b*192 + k
  int b = blk / 192, k = blk % 192;
  const float* src = (k < 64) ? x1 : (k < 128) ? x2 : x3;
  const float* p = src + ((size_t)b * 64 + (k & 63)) * 16384;
  float s = 0.f;
  for (int i = threadIdx.x; i < 16384; i += 256) s += p[i];
  __shared__ float red[256];
  red[threadIdx.x] = s;
  __syncthreads();
  for (int o = 128; o > 0; o >>= 1) {
    if (threadIdx.x < o) red[threadIdx.x] += red[threadIdx.x + o];
    __syncthreads();
  }
  if (threadIdx.x == 0) gap[blk] = red[0] * (1.f / 16384.f);
}

// -------- Kernel 2: fold BN into bf16 weights + scalar global branch --------
__global__ __launch_bounds__(256)
void prep_kernel(const float* __restrict__ gap,
    const float* lw1, const float* lb1, const float* lg1, const float* lbe1,
    const float* lm1, const float* lv1,
    const float* lw2, const float* lb2, const float* lg2, const float* lbe2,
    const float* lm2, const float* lv2,
    const float* gw1, const float* gb1, const float* gg1, const float* gbe1,
    const float* gm1, const float* gv1,
    const float* gw2, const float* gb2, const float* gg2, const float* gbe2,
    const float* gm2, const float* gv2,
    __bf16* __restrict__ w1f, float* __restrict__ b1f,
    __bf16* __restrict__ w2f, float* __restrict__ xgtot) {
  int tid = threadIdx.x;
  // Fold local conv1 + BN1:  w' = w * g/sqrt(v+eps);  b' = (b-m)*s + beta
  for (int i = tid; i < 96 * 192; i += 256) {
    int o = i / 192;
    float s = lg1[o] * rsqrtf(lv1[o] + EPS);
    w1f[i] = (__bf16)(lw1[i] * s);
  }
  for (int o = tid; o < 96; o += 256) {
    float s = lg1[o] * rsqrtf(lv1[o] + EPS);
    b1f[o] = (lb1[o] - lm1[o]) * s + lbe1[o];
  }
  // Fold local conv2 + BN2 (weights only; bias merged into xgtot below)
  for (int i = tid; i < 192 * 96; i += 256) {
    int o = i / 96;
    float s = lg2[o] * rsqrtf(lv2[o] + EPS);
    w2f[i] = (__bf16)(lw2[i] * s);
  }
  // Global branch in fp32 (tiny: 8 x (192*96 + 96*192) MACs)
  __shared__ float hgl[8 * 96];
  for (int i = tid; i < 8 * 96; i += 256) {
    int b = i / 96, o = i % 96;
    float dot = 0.f;
    const float* g = gap + b * 192;
    const float* w = gw1 + o * 192;
    for (int k = 0; k < 192; k++) dot += g[k] * w[k];
    float s = gg1[o] * rsqrtf(gv1[o] + EPS);
    float h = (dot + gb1[o] - gm1[o]) * s + gbe1[o];
    hgl[i] = h > 0.f ? h : 0.f;
  }
  __syncthreads();
  for (int i = tid; i < 8 * 192; i += 256) {
    int b = i / 192, ch = i % 192;
    float dot = 0.f;
    const float* h = hgl + b * 96;
    const float* w = gw2 + ch * 96;
    for (int k = 0; k < 96; k++) dot += h[k] * w[k];
    float s = gg2[ch] * rsqrtf(gv2[ch] + EPS);
    float xg = (dot + gb2[ch] - gm2[ch]) * s + gbe2[ch];
    float s2l = lg2[ch] * rsqrtf(lv2[ch] + EPS);
    float b2l = (lb2[ch] - lm2[ch]) * s2l + lbe2[ch];
    xgtot[i] = xg + b2l;   // per-(b,ch) additive term for the local logits
  }
}

// ----------- Kernel 3: per-pixel 192->96->192 convs via bf16 WMMA -----------
__global__ __launch_bounds__(256)
void mscam_kernel(const float* __restrict__ x1, const float* __restrict__ x2,
                  const float* __restrict__ x3,
                  const __bf16* __restrict__ w1f, const float* __restrict__ b1f,
                  const __bf16* __restrict__ w2f, const float* __restrict__ xgtot,
                  float* __restrict__ out) {
  __shared__ __align__(16) __bf16 Xt[128 * XSTRIDE];  // pixel-major concat x (bf16)
  __shared__ __align__(16) __bf16 Ht[128 * HSTRIDE];  // pixel-major hidden

  const int tid = threadIdx.x;
  const int b   = blockIdx.x >> 7;          // image
  const int hw0 = (blockIdx.x & 127) << 7;  // 128-pixel tile within H*W

  // Stage: coalesced global fp32 reads -> transposed bf16 LDS tile
  for (int i = tid; i < 192 * 128; i += 256) {
    int k = i >> 7, p = i & 127;
    const float* src = (k < 64) ? x1 : (k < 128) ? x2 : x3;
    float v = src[((size_t)b * 64 + (k & 63)) * 16384 + hw0 + p];
    Xt[p * XSTRIDE + k] = (__bf16)v;
  }
  __syncthreads();

  const int lane = tid & 31;
  const int ln   = lane & 15;   // A: M row within tile / B: N column
  const int half = lane >> 4;   // K sub-block selector
  const int m0   = (tid >> 5) * 16;  // wave-private pixel rows

  // ---- conv1: [16 pixels x 192] x [192 x 96] ----
  const __bf16* arow = &Xt[(m0 + ln) * XSTRIDE];
  v16bf afr[6];
#pragma unroll
  for (int kk = 0; kk < 6; kk++) afr[kk] = load_frag(arow, kk * 32 + half * 8);
#pragma unroll
  for (int n = 0; n < 6; n++) {
    v8f acc = {};
    const __bf16* bcol = w1f + (n * 16 + ln) * 192;
#pragma unroll
    for (int kk = 0; kk < 6; kk++)
      acc = __builtin_amdgcn_wmma_f32_16x16x32_bf16(
          false, afr[kk], false, load_frag(bcol, kk * 32 + half * 8),
          (short)0, acc, false, false);
    float bb = b1f[n * 16 + ln];
#pragma unroll
    for (int r = 0; r < 8; r++) {   // D layout: M = r + 8*half, N = ln
      float h = acc[r] + bb;
      h = h > 0.f ? h : 0.f;
      Ht[(m0 + r + 8 * half) * HSTRIDE + n * 16 + ln] = (__bf16)h;
    }
  }
  __syncthreads();

  // ---- conv2: [16 pixels x 96] x [96 x 192] ----
  const __bf16* hrow = &Ht[(m0 + ln) * HSTRIDE];
  v16bf hfr[3];
#pragma unroll
  for (int kk = 0; kk < 3; kk++) hfr[kk] = load_frag(hrow, kk * 32 + half * 8);
#pragma unroll
  for (int n = 0; n < 12; n++) {
    v8f acc = {};
    const __bf16* bcol = w2f + (n * 16 + ln) * 96;
#pragma unroll
    for (int kk = 0; kk < 3; kk++)
      acc = __builtin_amdgcn_wmma_f32_16x16x32_bf16(
          false, hfr[kk], false, load_frag(bcol, kk * 32 + half * 8),
          (short)0, acc, false, false);
    float zb = xgtot[b * 192 + n * 16 + ln];
#pragma unroll
    for (int r = 0; r < 8; r++)     // logits reuse the Xt buffer (wave-private rows)
      Xt[(m0 + r + 8 * half) * XSTRIDE + n * 16 + ln] = (__bf16)(acc[r] + zb);
  }
  __syncthreads();

  // ---- epilogue: sigmoid -> softmax over {c, c+64, c+128} -> weighted sum ----
  for (int it = 0; it < 32; it++) {
    int c = it * 2 + (tid >> 7);
    int p = tid & 127;
    float z0 = (float)Xt[p * XSTRIDE + c];
    float z1 = (float)Xt[p * XSTRIDE + c + 64];
    float z2 = (float)Xt[p * XSTRIDE + c + 128];
    float s0 = 1.f / (1.f + __expf(-z0));
    float s1 = 1.f / (1.f + __expf(-z1));
    float s2 = 1.f / (1.f + __expf(-z2));
    float mx = fmaxf(s0, fmaxf(s1, s2));
    float e0 = __expf(s0 - mx), e1 = __expf(s1 - mx), e2 = __expf(s2 - mx);
    float inv = 1.f / (e0 + e1 + e2);
    size_t gi = ((size_t)b * 64 + c) * 16384 + hw0 + p;  // coalesced (L2-hot)
    out[gi] = (e0 * x1[gi] + e1 * x2[gi] + e2 * x3[gi]) * inv;
  }
}

extern "C" void kernel_launch(void* const* d_in, const int* in_sizes, int n_in,
                              void* d_out, int out_size, void* d_ws, size_t ws_size,
                              hipStream_t stream) {
  const float* x1 = (const float*)d_in[0];
  const float* x2 = (const float*)d_in[1];
  const float* x3 = (const float*)d_in[2];
  const float* lp[12];
  const float* gp[12];
  for (int i = 0; i < 12; i++) lp[i] = (const float*)d_in[3 + i];
  for (int i = 0; i < 12; i++) gp[i] = (const float*)d_in[15 + i];

  char* ws = (char*)d_ws;
  float*  gap   = (float*)(ws + 0);        // 1536 f32
  float*  xgtot = (float*)(ws + 6144);     // 1536 f32
  float*  b1f   = (float*)(ws + 12288);    // 96 f32
  __bf16* w1f   = (__bf16*)(ws + 12800);   // 96*192 bf16
  __bf16* w2f   = (__bf16*)(ws + 49920);   // 192*96 bf16  (end ~86 KB)

  gap_kernel<<<dim3(8 * 192), dim3(256), 0, stream>>>(x1, x2, x3, gap);
  prep_kernel<<<dim3(1), dim3(256), 0, stream>>>(
      gap,
      lp[0], lp[1], lp[2], lp[3], lp[4], lp[5], lp[6], lp[7], lp[8], lp[9], lp[10], lp[11],
      gp[0], gp[1], gp[2], gp[3], gp[4], gp[5], gp[6], gp[7], gp[8], gp[9], gp[10], gp[11],
      w1f, b1f, w2f, xgtot);
  mscam_kernel<<<dim3(1024), dim3(256), 0, stream>>>(
      x1, x2, x3, w1f, b1f, w2f, xgtot, (float*)d_out);
}
